// RNNEncoder_57913339019644
// MI455X (gfx1250) — matrix operand order repaired and checked by reading, not compiled, not run on hardware
//
#include <hip/hip_runtime.h>

#define Bn 128
#define Tn 512
#define In 130
#define Hn 512
#define NWG 16
#define NTHREADS 256
#define WROWS 128          // gate rows per WG (32 cols x 4 gates)
#define WSTRIDE 648        // padded LDS row stride in bf16 elements (breaks bank conflicts)

typedef __attribute__((ext_vector_type(16))) __bf16 v16bf;
typedef __attribute__((ext_vector_type(8)))  float  v8f;
typedef __attribute__((ext_vector_type(4)))  unsigned int u32x4;
typedef __attribute__((ext_vector_type(2)))  float  f32x2;

union AFrag {
  v16bf v;
  u32x4 q[2];
  __bf16 e[16];
};

// ---- activations: prefer native CDNA5 V_TANH_F32 (TRANS unit) ----
#if __has_builtin(__builtin_amdgcn_tanhf)
__device__ __forceinline__ float tanh_n(float x) { return __builtin_amdgcn_tanhf(x); }
#define HAVE_NATIVE_TANH 1
#elif __has_builtin(__builtin_amdgcn_tanh_f32)
__device__ __forceinline__ float tanh_n(float x) { return __builtin_amdgcn_tanh_f32(x); }
#define HAVE_NATIVE_TANH 1
#else
#define HAVE_NATIVE_TANH 0
__device__ __forceinline__ float tanh_n(float x) {
  float e = __expf(-2.0f * fabsf(x));
  float r = (1.0f - e) / (1.0f + e);
  return copysignf(r, x);
}
#endif

__device__ __forceinline__ float sigmoid_n(float x) {
#if HAVE_NATIVE_TANH
  return __builtin_fmaf(0.5f, tanh_n(0.5f * x), 0.5f);
#else
  return 1.0f / (1.0f + __expf(-x));
#endif
}

__device__ __forceinline__ v8f vzero8() {
  v8f z;
#pragma unroll
  for (int r = 0; r < 8; ++r) z[r] = 0.0f;
  return z;
}

// Device-wide barrier for NWG persistent workgroups.
// Release: flush dirty WGP$ lines to device scope; Acquire: invalidate WGP$.
__device__ __forceinline__ void grid_barrier(unsigned* cnt, unsigned target) {
  asm volatile("global_wb scope:SCOPE_DEV" ::: "memory");
  asm volatile("s_wait_storecnt 0x0" ::: "memory");
  __syncthreads();
  if (threadIdx.x == 0) {
    __hip_atomic_fetch_add(cnt, 1u, __ATOMIC_RELEASE, __HIP_MEMORY_SCOPE_AGENT);
    while (__hip_atomic_load(cnt, __ATOMIC_ACQUIRE, __HIP_MEMORY_SCOPE_AGENT) < target) {
      __builtin_amdgcn_s_sleep(2);
    }
  }
  __syncthreads();
  asm volatile("global_inv scope:SCOPE_DEV" ::: "memory");
  asm volatile("s_wait_loadcnt 0x0" ::: "memory");
}

__global__ __launch_bounds__(NTHREADS, 1) void lstm_persistent_wmma(
    const float* __restrict__ inputs,   // (B,T,130)
    const int*   __restrict__ lengths,  // (B)
    const float* __restrict__ h0,       // (B,H)
    const float* __restrict__ c0,       // (B,H)
    const float* __restrict__ W_ih,     // (2048,128)
    const float* __restrict__ W_hh,     // (2048,512)
    const float* __restrict__ b_ih,     // (2048)
    const float* __restrict__ b_hh,     // (2048)
    float* __restrict__ out,            // (B,H)
    void* ws) {
  __shared__ __bf16 sW[WROWS * WSTRIDE];   // weight slice, bf16, [row_local][k 0..639]
  __shared__ float  sBias[WROWS];

  unsigned* cnt = (unsigned*)ws;
  __bf16* hbuf0 = (__bf16*)((char*)ws + 1024);
  __bf16* hbuf1 = hbuf0 + Bn * Hn;

  const int tid = threadIdx.x;
  const int wg = blockIdx.x;
  const int wgColBase = wg * 32;

  // ---- prologue: convert weight slice to bf16 in LDS ----
  // row_local = q2*64 + gate*16 + j  -> global gate row = gate*H + wgColBase + q2*16 + j
  for (int e = tid; e < WROWS * 640; e += NTHREADS) {
    int rl = e / 640;
    int k  = e - rl * 640;
    int g  = (rl >> 4) & 3;
    int j  = rl & 15;
    int q2 = rl >> 6;
    int grow = g * Hn + wgColBase + q2 * 16 + j;
    float v = (k < 128) ? W_ih[(size_t)grow * 128 + k]
                        : W_hh[(size_t)grow * Hn + (k - 128)];
    sW[rl * WSTRIDE + k] = (__bf16)v;
  }
  if (tid < WROWS) {
    int rl = tid;
    int g = (rl >> 4) & 3, j = rl & 15, q2 = rl >> 6;
    int grow = g * Hn + wgColBase + q2 * 16 + j;
    sBias[rl] = b_ih[grow] + b_hh[grow];
  }
  // init h buffer 0 from h0 (grid-strided, all WGs)
  for (int e = wg * NTHREADS + tid; e < Bn * Hn; e += NWG * NTHREADS) {
    hbuf0[e] = (__bf16)h0[e];
  }

  // ---- per-wave geometry ----
  const int wv    = tid >> 5;
  const int lane  = tid & 31;
  const int p     = wv & 3;        // M-pair index
  const int q     = wv >> 2;       // column quad (0/1): columns q*16..q*16+15
  const int mt0   = 2 * p;         // first global M tile
  const int lhalf = lane >> 4;     // 0 or 1
  const int ln    = lane & 15;
  const int col   = wgColBase + q * 16 + ln;   // global h column this lane owns

  // lengths + c0 into registers (c lives in VGPRs for all 512 steps)
  int   lenr[2][8];
  float creg[2][8];
#pragma unroll
  for (int mt = 0; mt < 2; ++mt) {
    int mbase = (mt0 + mt) * 16 + lhalf * 8;
#pragma unroll
    for (int r = 0; r < 8; ++r) {
      int m = mbase + r;
      lenr[mt][r] = lengths[m];
      creg[mt][r] = c0[(size_t)m * Hn + col];
    }
  }

  unsigned phase = 1;
  grid_barrier(cnt, phase * NWG); phase++;

  // B-fragment LDS row bases (one per gate); includes lane-half K offset
  const __bf16* wrow[4];
#pragma unroll
  for (int g = 0; g < 4; ++g)
    wrow[g] = sW + (size_t)(q * 64 + g * 16 + ln) * WSTRIDE + lhalf * 8;

  // loop-invariant biases for this lane's column
  const float bi  = sBias[q * 64 +  0 + ln];
  const float bff = sBias[q * 64 + 16 + ln];
  const float bg  = sBias[q * 64 + 32 + ln];
  const float bo  = sBias[q * 64 + 48 + ln];

  for (int t = 0; t < Tn; ++t) {
    const __bf16* hread  = (t & 1) ? hbuf1 : hbuf0;
    __bf16*       hwrite = (t & 1) ? hbuf0 : hbuf1;

    v8f acc[2][4];
#pragma unroll
    for (int mt = 0; mt < 2; ++mt)
#pragma unroll
      for (int g = 0; g < 4; ++g) acc[mt][g] = vzero8();

    // ---- K region 1: x_t (K = 0..127), fp32 -> bf16 on the fly ----
    for (int ks = 0; ks < 4; ++ks) {
      int k0 = ks * 32;
      AFrag a[2];
#pragma unroll
      for (int mt = 0; mt < 2; ++mt) {
        int m = (mt0 + mt) * 16 + ln;
        const float* xr = inputs + ((size_t)m * Tn + t) * In + k0 + lhalf * 8;
        f32x2 f[8];
        f[0] = *(const f32x2*)(xr + 0);
        f[1] = *(const f32x2*)(xr + 2);
        f[2] = *(const f32x2*)(xr + 4);
        f[3] = *(const f32x2*)(xr + 6);
        f[4] = *(const f32x2*)(xr + 16);
        f[5] = *(const f32x2*)(xr + 18);
        f[6] = *(const f32x2*)(xr + 20);
        f[7] = *(const f32x2*)(xr + 22);
#pragma unroll
        for (int j = 0; j < 8; ++j) {
          a[mt].e[2 * j]     = (__bf16)f[j][0];
          a[mt].e[2 * j + 1] = (__bf16)f[j][1];
        }
      }
#pragma unroll
      for (int g = 0; g < 4; ++g) {
        AFrag b;
        const __bf16* wp = wrow[g] + k0;
        b.q[0] = *(const u32x4*)(wp);
        b.q[1] = *(const u32x4*)(wp + 16);
#pragma unroll
        for (int mt = 0; mt < 2; ++mt)
          acc[mt][g] = __builtin_amdgcn_wmma_f32_16x16x32_bf16(
              false, a[mt].v, false, b.v, (short)0, acc[mt][g], false, false);
      }
    }

    // ---- K region 2: h (K = 128..639), bf16 direct ----
    for (int ks = 0; ks < 16; ++ks) {
      int hk0 = ks * 32;
      AFrag a[2];
#pragma unroll
      for (int mt = 0; mt < 2; ++mt) {
        int m = (mt0 + mt) * 16 + ln;
        const __bf16* hr = hread + (size_t)m * Hn + hk0 + lhalf * 8;
        a[mt].q[0] = *(const u32x4*)(hr);
        a[mt].q[1] = *(const u32x4*)(hr + 16);
      }
      int k0 = 128 + hk0;
#pragma unroll
      for (int g = 0; g < 4; ++g) {
        AFrag b;
        const __bf16* wp = wrow[g] + k0;
        b.q[0] = *(const u32x4*)(wp);
        b.q[1] = *(const u32x4*)(wp + 16);
#pragma unroll
        for (int mt = 0; mt < 2; ++mt)
          acc[mt][g] = __builtin_amdgcn_wmma_f32_16x16x32_bf16(
              false, a[mt].v, false, b.v, (short)0, acc[mt][g], false, false);
      }
    }

    // ---- epilogue: gates -> c,h update (register-resident c) ----
    int tp1 = t + 1;
#pragma unroll
    for (int mt = 0; mt < 2; ++mt) {
      int mbase = (mt0 + mt) * 16 + lhalf * 8;
#pragma unroll
      for (int r = 0; r < 8; ++r) {
        float ig = sigmoid_n(acc[mt][0][r] + bi);
        float fg = sigmoid_n(acc[mt][1][r] + bff);
        float gg = tanh_n(acc[mt][2][r] + bg);
        float og = sigmoid_n(acc[mt][3][r] + bo);
        float cn = fg * creg[mt][r] + ig * gg;
        creg[mt][r] = cn;
        float hn = og * tanh_n(cn);
        int m = mbase + r;
        hwrite[(size_t)m * Hn + col] = (__bf16)hn;
        if (lenr[mt][r] == tp1) out[(size_t)m * Hn + col] = hn;
      }
    }

    grid_barrier(cnt, phase * NWG); phase++;
  }
}

extern "C" void kernel_launch(void* const* d_in, const int* in_sizes, int n_in,
                              void* d_out, int out_size, void* d_ws, size_t ws_size,
                              hipStream_t stream) {
  (void)in_sizes; (void)n_in; (void)out_size; (void)ws_size;
  const float* inputs  = (const float*)d_in[0];
  const int*   lengths = (const int*)d_in[1];
  const float* h0      = (const float*)d_in[2];
  const float* c0      = (const float*)d_in[3];
  const float* W_ih    = (const float*)d_in[4];
  const float* W_hh    = (const float*)d_in[5];
  const float* b_ih    = (const float*)d_in[6];
  const float* b_hh    = (const float*)d_in[7];
  float* out = (float*)d_out;

  // ws layout: [0,4) barrier counter (zeroed each call), [1024, +128KB) h buf0,
  // then h buf1. Counter must be 0 at kernel start; memset is capture-safe.
  hipMemsetAsync(d_ws, 0, 1024, stream);

  lstm_persistent_wmma<<<dim3(NWG), dim3(NTHREADS), 0, stream>>>(
      inputs, lengths, h0, c0, W_ih, W_hh, b_ih, b_hh, out, d_ws);
}
